// customREDnetSparseObjectDet_21071109554421
// MI455X (gfx1250) — matrix operand order repaired and checked by reading, not compiled, not run on hardware
//
#include <hip/hip_runtime.h>
#include <hip/hip_bf16.h>

typedef __attribute__((ext_vector_type(16))) _Float16 v16h;
typedef __attribute__((ext_vector_type(8)))  float    v8f;

#define BLK 256
#define EPSBN 1e-4f

#if defined(__AMDGCN__) && __has_builtin(__builtin_amdgcn_global_load_async_to_lds_b128)
#define ASYNC_LDS 1
typedef __attribute__((__vector_size__(4 * sizeof(int)))) int async_v4i;
#define GPTR(p) ((__attribute__((address_space(1))) async_v4i*)(p))
#define LPTR(p) ((__attribute__((address_space(3))) async_v4i*)(p))
#else
#define ASYNC_LDS 0
#endif

static __device__ __forceinline__ void wait_async_lds() {
#if ASYNC_LDS
#if __has_builtin(__builtin_amdgcn_s_wait_asynccnt)
    __builtin_amdgcn_s_wait_asynccnt(0);
#else
    asm volatile("s_wait_asynccnt 0x0" ::: "memory");
#endif
#endif
}

static __device__ __forceinline__ float sigm(float x) { return 1.0f / (1.0f + expf(-x)); }

// ---------------------------------------------------------------------------
// int mask -> float
__global__ void i2f_kernel(const int* __restrict__ src, float* __restrict__ dst, long long total) {
    for (long long i = blockIdx.x * (long long)blockDim.x + threadIdx.x; i < total;
         i += (long long)gridDim.x * blockDim.x)
        dst[i] = (float)src[i];
}

// f32 [K,N] -> f16 [K,Np] (row-major, zero-pad columns >= N)   [used for FC1 act]
__global__ void f32_to_f16_pad(const float* __restrict__ src, _Float16* __restrict__ dst,
                               int K, int N, int Np, long long total) {
    for (long long i = blockIdx.x * (long long)blockDim.x + threadIdx.x; i < total;
         i += (long long)gridDim.x * blockDim.x) {
        int n = (int)(i % Np);
        long long k = i / Np;
        dst[i] = (n < N) ? (_Float16)src[k * N + n] : (_Float16)0.0f;
    }
}

// f32 [K,N] -> f16 transposed [Np,K] (zero-pad rows >= N)       [B operands]
__global__ void f32_to_f16_T(const float* __restrict__ src, _Float16* __restrict__ dst,
                             int K, int N, int Np, long long total) {
    for (long long i = blockIdx.x * (long long)blockDim.x + threadIdx.x; i < total;
         i += (long long)gridDim.x * blockDim.x) {
        int k = (int)(i % K);
        int n = (int)(i / K);
        dst[i] = (n < N) ? (_Float16)src[(long long)k * N + n] : (_Float16)0.0f;
    }
}

// ---------------------------------------------------------------------------
// Direct masked SAME 3x3 conv (small-Cin layers 1..3).
__global__ void conv3x3_masked(const float* __restrict__ in, const float* __restrict__ Wt,
                               const float* __restrict__ mask, float* __restrict__ out,
                               int NIMG, int H, int Wd, int Ci, int Co, int maskInput,
                               long long total) {
    for (long long idx = blockIdx.x * (long long)blockDim.x + threadIdx.x; idx < total;
         idx += (long long)gridDim.x * blockDim.x) {
        int co = (int)(idx % Co);
        long long p = idx / Co;
        int x = (int)(p % Wd);
        long long t = p / Wd;
        int y = (int)(t % H);
        int n = (int)(t / H);
        float m = mask[p];
        if (m == 0.0f) { out[idx] = 0.0f; continue; }
        float acc = 0.0f;
        for (int dy = -1; dy <= 1; ++dy) {
            int yy = y + dy; if (yy < 0 || yy >= H) continue;
            for (int dx = -1; dx <= 1; ++dx) {
                int xx = x + dx; if (xx < 0 || xx >= Wd) continue;
                long long pin = ((long long)n * H + yy) * Wd + xx;
                float mm = 1.0f;
                if (maskInput) { mm = mask[pin]; if (mm == 0.0f) continue; }
                const float* ip = in + pin * Ci;
                const float* wp = Wt + (((dy + 1) * 3) + (dx + 1)) * Ci * Co + co;
                float s = 0.0f;
                for (int ci = 0; ci < Ci; ++ci) s += ip[ci] * wp[ci * Co];
                acc += s * mm;
            }
        }
        out[idx] = acc;
    }
}

// ---------------------------------------------------------------------------
// n = sum(mask)
__global__ void mask_count(const float* __restrict__ mask, long long P, float* out) {
    __shared__ float ss[BLK];
    int tid = threadIdx.x;
    float s = 0.0f;
    for (long long i = blockIdx.x * (long long)blockDim.x + tid; i < P;
         i += (long long)gridDim.x * blockDim.x)
        s += mask[i];
    ss[tid] = s;
    __syncthreads();
    for (int st = BLK / 2; st > 0; st >>= 1) {
        if (tid < st) ss[tid] += ss[tid + st];
        __syncthreads();
    }
    if (tid == 0) atomicAdd(out, ss[0]);
}

// Per-channel sum / sumsq over active sites (masked entries contribute 0).
__global__ void bn_stats(const float* __restrict__ x, const float* __restrict__ maskOrNull,
                         float* __restrict__ stats, long long P, int C) {
    __shared__ float ss[256];
    int tid = threadIdx.x;
    for (int i = tid; i < 2 * C; i += blockDim.x) ss[i] = 0.0f;
    __syncthreads();
    long long total = P * C;
    for (long long i = blockIdx.x * (long long)blockDim.x + tid; i < total;
         i += (long long)gridDim.x * blockDim.x) {
        int c = (int)(i % C);
        long long p = i / C;
        float v = x[i];
        if (maskOrNull) v *= maskOrNull[p];
        atomicAdd(&ss[c], v);
        atomicAdd(&ss[C + c], v * v);
    }
    __syncthreads();
    for (int i = tid; i < 2 * C; i += blockDim.x) atomicAdd(&stats[i], ss[i]);
}

// In-place BN + ReLU + mask
__global__ void bn_apply(float* __restrict__ x, const float* __restrict__ mask,
                         const float* __restrict__ stats, const float* __restrict__ nPtr,
                         const float* __restrict__ g, const float* __restrict__ b,
                         long long P, int C) {
    float n = fmaxf(nPtr[0], 1.0f);
    long long total = P * C;
    for (long long i = blockIdx.x * (long long)blockDim.x + threadIdx.x; i < total;
         i += (long long)gridDim.x * blockDim.x) {
        int c = (int)(i % C);
        long long p = i / C;
        float mean = stats[c] / n;
        float var = stats[C + c] / n - mean * mean;
        float y = (x[i] - mean) * rsqrtf(var + EPSBN) * g[c] + b[c];
        x[i] = fmaxf(y, 0.0f) * mask[p];
    }
}

// ---------------------------------------------------------------------------
// 3x3 / stride-2 VALID masked maxpool; output active if any input active
__global__ void maxpool3x3s2(const float* __restrict__ in, const float* __restrict__ maskIn,
                             float* __restrict__ out, float* __restrict__ maskOut,
                             int NIMG, int Hi, int Wi, int C, int Ho, int Wo, long long total) {
    for (long long idx = blockIdx.x * (long long)blockDim.x + threadIdx.x; idx < total;
         idx += (long long)gridDim.x * blockDim.x) {
        int c = (int)(idx % C);
        long long p = idx / C;
        int xo = (int)(p % Wo);
        long long t = p / Wo;
        int yo = (int)(t % Ho);
        int n = (int)(t / Ho);
        float best = -1e30f, mm = 0.0f;
        for (int ky = 0; ky < 3; ++ky) {
            int yi = yo * 2 + ky; if (yi >= Hi) continue;
            for (int kx = 0; kx < 3; ++kx) {
                int xi = xo * 2 + kx; if (xi >= Wi) continue;
                long long pi = ((long long)n * Hi + yi) * Wi + xi;
                float mv = maskIn[pi];
                mm = fmaxf(mm, mv);
                float xv = (mv > 0.0f) ? in[pi * C + c] : -1e30f;
                best = fmaxf(best, xv);
            }
        }
        out[idx] = (mm > 0.0f) ? best : 0.0f;
        if (c == 0) maskOut[p] = mm;
    }
}

// ---------------------------------------------------------------------------
// im2col for SAME 3x3: out[p, (dy*3+dx)*C + ci] as f16; zero outside image
__global__ void im2col3x3(const float* __restrict__ in, _Float16* __restrict__ out,
                          int NIMG, int H, int Wd, int C, long long total) {
    int KC = 9 * C;
    for (long long idx = blockIdx.x * (long long)blockDim.x + threadIdx.x; idx < total;
         idx += (long long)gridDim.x * blockDim.x) {
        int k = (int)(idx % KC);
        long long p = idx / KC;
        int ci = k % C;
        int tap = k / C;
        int dy = tap / 3 - 1, dx = tap % 3 - 1;
        int x = (int)(p % Wd);
        long long t = p / Wd;
        int y = (int)(t % H);
        int n = (int)(t / H);
        int yy = y + dy, xx = x + dx;
        float v = 0.0f;
        if (yy >= 0 && yy < H && xx >= 0 && xx < Wd)
            v = in[(((long long)n * H + yy) * Wd + xx) * C + ci];
        out[idx] = (_Float16)v;
    }
}

// ---------------------------------------------------------------------------
// WMMA GEMM: C[M,Nstore] = A[M,K](f16, row-major) x BT[Np,K](f16, row-major = B^T)
// Block: 8 waves -> 128(M) x 64(N) tile; wave -> 32x32 (4 accumulators, 4 WMMA/K-step).
// Requires M%128==0, K%32==0, grid.y*64 == Np. Double-buffered LDS staging via
// GLOBAL_LOAD_ASYNC_TO_LDS_B128 (ASYNCcnt) when available.
__global__ void __launch_bounds__(256) gemm_wmma_f16(
    const _Float16* __restrict__ A, const _Float16* __restrict__ BT,
    float* __restrict__ C, int M, int Nstore, int K, int ldc,
    const float* __restrict__ bias, int relu) {
    __shared__ _Float16 sA[2][128 * 32];   // [m][k]
    __shared__ _Float16 sB[2][64 * 32];    // [n][k]
    const int tid  = threadIdx.x;
    const int wave = tid >> 5;
    const int lane = tid & 31;
    const int wm = wave & 3;               // 4 waves along M (32 rows each)
    const int wn = wave >> 2;              // 2 waves along N (32 cols each)
    const int mBase = blockIdx.x * 128;
    const int nBase = blockIdx.y * 64;

    v8f acc[2][2];
#pragma unroll
    for (int i = 0; i < 2; ++i)
#pragma unroll
        for (int j = 0; j < 2; ++j)
            acc[i][j] = (v8f){0.f, 0.f, 0.f, 0.f, 0.f, 0.f, 0.f, 0.f};

    // tile staging: A 128x32 halves (512 x b128), B 64x32 halves (256 x b128)
    auto stage = [&](int buf, int k0) {
#pragma unroll
        for (int i = 0; i < 2; ++i) {
            int idx = tid + i * 256;       // 0..511
            int row = idx >> 2;
            int c8 = (idx & 3) << 3;       // halves
            const _Float16* gp = A + (size_t)(mBase + row) * K + k0 + c8;
            _Float16* lp = &sA[buf][row * 32 + c8];
#if ASYNC_LDS
            __builtin_amdgcn_global_load_async_to_lds_b128(GPTR(gp), LPTR(lp), 0, 0);
#else
            *reinterpret_cast<uint4*>(lp) = *reinterpret_cast<const uint4*>(gp);
#endif
        }
        {
            int row = tid >> 2;            // 0..63
            int c8 = (tid & 3) << 3;
            const _Float16* gp = BT + (size_t)(nBase + row) * K + k0 + c8;
            _Float16* lp = &sB[buf][row * 32 + c8];
#if ASYNC_LDS
            __builtin_amdgcn_global_load_async_to_lds_b128(GPTR(gp), LPTR(lp), 0, 0);
#else
            *reinterpret_cast<uint4*>(lp) = *reinterpret_cast<const uint4*>(gp);
#endif
        }
    };

    stage(0, 0);
    wait_async_lds();
    __syncthreads();

    const int mrow = lane & 15;            // row (A) / col (B) within 16x16
    const int kh = lane >> 4;
    int buf = 0;
    for (int k0 = 0; k0 < K; k0 += 32) {
        if (k0 + 32 < K) stage(buf ^ 1, k0 + 32);

        // gather fragments (ISA 7.12.2 16-bit layouts); pairs merge to ds_load_b128
        union Frag { v16h v; unsigned u[8]; } fa[2], fb[2];
#pragma unroll
        for (int t = 0; t < 2; ++t) {
            const _Float16* pA = &sA[buf][(wm * 32 + t * 16 + mrow) * 32];
            const _Float16* pB = &sB[buf][(wn * 32 + t * 16 + mrow) * 32];
#pragma unroll
            for (int e2 = 0; e2 < 8; ++e2) {
                int e = e2 << 1;
                int kA = ((e >> 3) << 4) + (kh << 3) + (((e >> 1) & 3) << 1);
                fa[t].u[e2] = *reinterpret_cast<const unsigned*>(pA + kA);
                int kB = ((e >> 3) << 4) + (kh << 3) + (((e >> 1) & 3) << 1);
                fb[t].u[e2] = *reinterpret_cast<const unsigned*>(pB + kB);
            }
        }
#pragma unroll
        for (int am = 0; am < 2; ++am)
#pragma unroll
            for (int bn = 0; bn < 2; ++bn)
                acc[am][bn] = __builtin_amdgcn_wmma_f32_16x16x32_f16(
                    false, fa[am].v, false, fb[bn].v, (short)0, acc[am][bn], false, false);

        if (k0 + 32 < K) wait_async_lds();
        __syncthreads();
        buf ^= 1;
    }

    // C/D layout: VGPR r -> M = r + 8*(lane>>4); N = lane&15
    const int ncol = lane & 15;
    const int mhi = (lane >> 4) << 3;
#pragma unroll
    for (int am = 0; am < 2; ++am) {
#pragma unroll
        for (int bn = 0; bn < 2; ++bn) {
#pragma unroll
            for (int r = 0; r < 8; ++r) {
                int gm = mBase + wm * 32 + am * 16 + mhi + r;
                int gn = nBase + wn * 32 + bn * 16 + ncol;
                if (gn < Nstore) {
                    float v = acc[am][bn][r];
                    if (bias) v += bias[gn];
                    if (relu) v = fmaxf(v, 0.0f);
                    C[(size_t)gm * ldc + gn] = v;
                }
            }
        }
    }
}

// ---------------------------------------------------------------------------
// LSTM gate nonlinearity + mask + channel-major flatten -> f16 A for FC1.
__global__ void lstm_act(const float* __restrict__ gates, const float* __restrict__ mask,
                         _Float16* __restrict__ flatA, long long P, long long total) {
    for (long long idx = blockIdx.x * (long long)blockDim.x + threadIdx.x; idx < total;
         idx += (long long)gridDim.x * blockDim.x) {
        int c = (int)(idx % 256);
        long long p = idx / 256;
        const float* gp = gates + p * 1024;
        float iv = gp[c], ov = gp[512 + c], gv = gp[768 + c];
        float cc = sigm(iv) * tanhf(gv);
        float hh = sigm(ov) * tanhf(cc) * mask[p];
        long long b = p / 35;
        int hw = (int)(p % 35);
        flatA[b * 8960 + (long long)c * 35 + hw] = (_Float16)hh;
    }
}

// ---------------------------------------------------------------------------
static inline int nblk(long long total) {
    long long b = (total + BLK - 1) / BLK;
    if (b > (1 << 20)) b = 1 << 20;
    return (int)b;
}

extern "C" void kernel_launch(void* const* d_in, const int* in_sizes, int n_in,
                              void* d_out, int out_size, void* d_ws, size_t ws_size,
                              hipStream_t stream) {
    const float* x   = (const float*)d_in[0];
    const int*   msk = (const int*)d_in[1];
    const float* W1  = (const float*)d_in[2];
    const float* g1  = (const float*)d_in[3];
    const float* b1  = (const float*)d_in[4];
    const float* W2  = (const float*)d_in[5];
    const float* g2  = (const float*)d_in[6];
    const float* b2  = (const float*)d_in[7];
    const float* W3  = (const float*)d_in[8];
    const float* g3  = (const float*)d_in[9];
    const float* b3  = (const float*)d_in[10];
    const float* W4  = (const float*)d_in[11];
    const float* g4  = (const float*)d_in[12];
    const float* b4  = (const float*)d_in[13];
    const float* Wl  = (const float*)d_in[14];
    const float* bl  = (const float*)d_in[15];
    const float* lw1 = (const float*)d_in[16];
    const float* lb1 = (const float*)d_in[17];
    const float* lw2 = (const float*)d_in[18];
    const float* lb2 = (const float*)d_in[19];
    float* out = (float*)d_out;

    const int B = 2048, H1 = 23, Wd1 = 31;
    const long long P1 = (long long)B * H1 * Wd1;          // 1,460,224
    const int H2 = 11, Wd2 = 15;
    const long long P2 = (long long)B * H2 * Wd2;          // 337,920
    const int H3 = 5, Wd3 = 7;
    const long long P3 = (long long)B * H3 * Wd3;          // 71,680

    // ---- workspace layout (lifetime-overlaid arenas) ----
    char* ws = (char*)d_ws;
    size_t off = 0;
    auto alloc = [&](size_t bytes) -> char* {
        char* p = ws + off;
        off += (bytes + 255) & ~(size_t)255;
        return p;
    };
    float*    maskF = (float*)alloc(P1 * 4);
    float*    stats = (float*)alloc(2048);                 // s1[32] s2[64] s3[128] s4[256] n1 n2
    char*     arenaA = alloc(389283840);                   // c2out | im2col4 | gates
    char*     arenaB = alloc(173015040);                   // c1out | c4out | flatA+fc1+fc1h
    char*     arenaC = alloc(165150720);                   // c3out | im2colL
    float*    h2p   = (float*)alloc(P2 * 32 * 4);
    float*    mask1 = (float*)alloc(P2 * 4);
    float*    h4p   = (float*)alloc(P3 * 128 * 4);
    float*    mask2 = (float*)alloc(P3 * 4);
    _Float16* W4t   = (_Float16*)alloc((size_t)128 * 576 * 2);
    _Float16* Wlt   = (_Float16*)alloc((size_t)1024 * 1152 * 2);
    _Float16* lw1t  = (_Float16*)alloc((size_t)1024 * 8960 * 2);
    _Float16* lw2t  = (_Float16*)alloc((size_t)448 * 1024 * 2);

    float* s1 = stats;        float* s2 = stats + 32;
    float* s3 = stats + 96;   float* s4 = stats + 224;
    float* n1 = stats + 480;  float* n2 = stats + 481;

    float*    c1out   = (float*)arenaB;
    float*    c2out   = (float*)arenaA;
    float*    c3out   = (float*)arenaC;
    _Float16* im2col4 = (_Float16*)arenaA;                 // after c2out dead
    float*    c4out   = (float*)arenaB;                    // after c1out dead
    _Float16* im2colL = (_Float16*)arenaC;                 // after c3out dead
    float*    gates   = (float*)arenaA;                    // after im2col4 dead
    _Float16* flatA   = (_Float16*)arenaB;                 // after c4out dead
    float*    fc1     = (float*)(arenaB + 36700160);
    _Float16* fc1h    = (_Float16*)(arenaB + 45088768);

    (void)hipMemsetAsync(stats, 0, 2048, stream);

    // mask -> float; weight conversions to f16 (B operands stored transposed [N,K])
    i2f_kernel<<<nblk(P1), BLK, 0, stream>>>(msk, maskF, P1);
    f32_to_f16_T<<<nblk((long long)128 * 576), BLK, 0, stream>>>(W4, W4t, 576, 128, 128, (long long)128 * 576);
    f32_to_f16_T<<<nblk((long long)1024 * 1152), BLK, 0, stream>>>(Wl, Wlt, 1152, 1024, 1024, (long long)1024 * 1152);
    f32_to_f16_T<<<nblk((long long)1024 * 8960), BLK, 0, stream>>>(lw1, lw1t, 8960, 1024, 1024, (long long)1024 * 8960);
    f32_to_f16_T<<<nblk((long long)448 * 1024), BLK, 0, stream>>>(lw2, lw2t, 1024, 420, 448, (long long)448 * 1024);
    mask_count<<<nblk(P1), BLK, 0, stream>>>(maskF, P1, n1);

    // conv1 (2->16) + BN1
    conv3x3_masked<<<nblk(P1 * 16), BLK, 0, stream>>>(x, W1, maskF, c1out, B, H1, Wd1, 2, 16, 1, P1 * 16);
    bn_stats<<<nblk(P1 * 16), BLK, 0, stream>>>(c1out, nullptr, s1, P1, 16);
    bn_apply<<<nblk(P1 * 16), BLK, 0, stream>>>(c1out, maskF, s1, n1, g1, b1, P1, 16);

    // conv2 (16->32) + BN2
    conv3x3_masked<<<nblk(P1 * 32), BLK, 0, stream>>>(c1out, W2, maskF, c2out, B, H1, Wd1, 16, 32, 0, P1 * 32);
    bn_stats<<<nblk(P1 * 32), BLK, 0, stream>>>(c2out, nullptr, s2, P1, 32);
    bn_apply<<<nblk(P1 * 32), BLK, 0, stream>>>(c2out, maskF, s2, n1, g2, b2, P1, 32);

    // pool1 -> [B,11,15,32]
    maxpool3x3s2<<<nblk(P2 * 32), BLK, 0, stream>>>(c2out, maskF, h2p, mask1, B, H1, Wd1, 32, H2, Wd2, P2 * 32);
    mask_count<<<nblk(P2), BLK, 0, stream>>>(mask1, P2, n2);

    // conv3 (32->64) + BN3
    conv3x3_masked<<<nblk(P2 * 64), BLK, 0, stream>>>(h2p, W3, mask1, c3out, B, H2, Wd2, 32, 64, 0, P2 * 64);
    bn_stats<<<nblk(P2 * 64), BLK, 0, stream>>>(c3out, nullptr, s3, P2, 64);
    bn_apply<<<nblk(P2 * 64), BLK, 0, stream>>>(c3out, mask1, s3, n2, g3, b3, P2, 64);

    // conv4 (64->128) via im2col + WMMA GEMM; BN4 stats masked (GEMM is unmasked)
    im2col3x3<<<nblk(P2 * 576), BLK, 0, stream>>>(c3out, im2col4, B, H2, Wd2, 64, P2 * 576);
    {
        dim3 grid((unsigned)(P2 / 128), 2);                // 2640 x 2
        gemm_wmma_f16<<<grid, 256, 0, stream>>>(im2col4, W4t, c4out, (int)P2, 128, 576, 128, nullptr, 0);
    }
    bn_stats<<<nblk(P2 * 128), BLK, 0, stream>>>(c4out, mask1, s4, P2, 128);
    bn_apply<<<nblk(P2 * 128), BLK, 0, stream>>>(c4out, mask1, s4, n2, g4, b4, P2, 128);

    // pool2 -> [B,5,7,128]
    maxpool3x3s2<<<nblk(P3 * 128), BLK, 0, stream>>>(c4out, mask1, h4p, mask2, B, H2, Wd2, 128, H3, Wd3, P3 * 128);

    // ConvLSTM gate conv (128->1024) via im2col + WMMA GEMM (+bias bl)
    im2col3x3<<<nblk(P3 * 1152), BLK, 0, stream>>>(h4p, im2colL, B, H3, Wd3, 128, P3 * 1152);
    {
        dim3 grid((unsigned)(P3 / 128), 16);               // 560 x 16
        gemm_wmma_f16<<<grid, 256, 0, stream>>>(im2colL, Wlt, gates, (int)P3, 1024, 1152, 1024, bl, 0);
    }
    // gate nonlinearity + mask + channel-major flatten -> flatA f16 [2048, 8960]
    lstm_act<<<nblk(P3 * 256), BLK, 0, stream>>>(gates, mask2, flatA, P3, P3 * 256);

    // FC1: [2048,8960] x [8960,1024] + bias + ReLU (WMMA), then f32->f16
    {
        dim3 grid(16, 16);
        gemm_wmma_f16<<<grid, 256, 0, stream>>>(flatA, lw1t, fc1, 2048, 1024, 8960, 1024, lb1, 1);
    }
    f32_to_f16_pad<<<nblk((long long)2048 * 1024), BLK, 0, stream>>>(fc1, fc1h, 2048, 1024, 1024, (long long)2048 * 1024);

    // FC2: [2048,1024] x [1024,420] + bias -> d_out (N padded to 448, guarded store)
    {
        dim3 grid(16, 7);
        gemm_wmma_f16<<<grid, 256, 0, stream>>>(fc1h, lw2t, out, 2048, 420, 1024, 420, lb2, 0);
    }

    (void)in_sizes; (void)n_in; (void)out_size; (void)ws_size;
}
